// MambaDFuse_25168508354828
// MI455X (gfx1250) — compile-verified, matrix operand order checked by reading, not run on hardware
//
#include <hip/hip_runtime.h>
#include <hip/hip_bf16.h>
#include <math.h>

// ---------------------------------------------------------------------------
// Types for CDNA5 WMMA (wave32, 16x16x32 bf16 -> f32)
// ---------------------------------------------------------------------------
typedef __attribute__((ext_vector_type(16))) __bf16 v16bf;
typedef __attribute__((ext_vector_type(8)))  __bf16 v8bf;
typedef __attribute__((ext_vector_type(8)))  float  v8f;

__device__ __forceinline__ float silu_f(float x) { return x / (1.0f + __expf(-x)); }

static inline int rup32(int k) { return (k + 31) & ~31; }
static inline int rup64(int n) { return (n + 63) & ~63; }

__device__ __forceinline__ v16bf cat16(v8bf lo, v8bf hi)
{
    return __builtin_shufflevector(lo, hi, 0,1,2,3,4,5,6,7,8,9,10,11,12,13,14,15);
}

// ---------------------------------------------------------------------------
// GEMM:  C[M,N] (fp32, row-major ldc) = act( Abf[M,Kp] * Wt^T + bias )
//   Abf : bf16, row-major, rows padded/zero-filled to Kp (multiple of 32)
//   Wt  : bf16, N-major (Npad rows of Kp), zero-padded in both dims
// 4 waves/block; wave w owns a 32(M) x 64(N) slice of a 128x64 block tile:
// 2 A fragments share 4 B fragments -> 8 WMMA per K-step on 12 16B loads.
// act: 0 = none, 1 = softplus.  accum: add into existing C (residual).
// ---------------------------------------------------------------------------
__global__ void __launch_bounds__(128)
gemm_wmma_bf16(const __bf16* __restrict__ Abf,
               const __bf16* __restrict__ Wt,
               float* __restrict__ C, int ldc,
               const float* __restrict__ bias,
               int M, int N, int Kp, int act, int accum)
{
    const int lane = threadIdx.x & 31;
    const int wav  = threadIdx.x >> 5;     // 0..3
    const int hsel = lane >> 4;            // 0: lanes 0-15, 1: lanes 16-31
    const int lr   = lane & 15;

    const int tile_m = blockIdx.y * 128 + wav * 32;
    const int tn0    = blockIdx.x * 64;

    const int mr0 = ((tile_m + lr)      < M) ? (tile_m + lr)      : 0;  // safety clamp
    const int mr1 = ((tile_m + 16 + lr) < M) ? (tile_m + 16 + lr) : 0;

    // Fragment base pointers (16B aligned: Kp multiple of 32 bf16 = 64B rows)
    const __bf16* arow0 = Abf + (size_t)mr0 * Kp + 8 * hsel;
    const __bf16* arow1 = Abf + (size_t)mr1 * Kp + 8 * hsel;
    const __bf16* brow0 = Wt + (size_t)(tn0 + lr)      * Kp + 8 * hsel;
    const __bf16* brow1 = Wt + (size_t)(tn0 + 16 + lr) * Kp + 8 * hsel;
    const __bf16* brow2 = Wt + (size_t)(tn0 + 32 + lr) * Kp + 8 * hsel;
    const __bf16* brow3 = Wt + (size_t)(tn0 + 48 + lr) * Kp + 8 * hsel;

    v8f acc0[4], acc1[4];
    #pragma unroll
    for (int t = 0; t < 4; ++t) {
        acc0[t] = (v8f){0,0,0,0,0,0,0,0};
        acc1[t] = (v8f){0,0,0,0,0,0,0,0};
    }

    const bool full = (tile_m + 32 <= M) && (tn0 + 64 <= N);

    if (accum) {
        if (full) {
            #pragma unroll
            for (int t = 0; t < 4; ++t) {
                const int nc = tn0 + 16 * t + lr;
                #pragma unroll
                for (int r = 0; r < 8; ++r) {
                    acc0[t][r] = C[(size_t)(tile_m + r + 8 * hsel) * ldc + nc];
                    acc1[t][r] = C[(size_t)(tile_m + 16 + r + 8 * hsel) * ldc + nc];
                }
            }
        } else {
            #pragma unroll
            for (int t = 0; t < 4; ++t) {
                const int nc = tn0 + 16 * t + lr;
                #pragma unroll
                for (int r = 0; r < 8; ++r) {
                    const int m0 = tile_m + r + 8 * hsel;
                    const int m1 = m0 + 16;
                    if (m0 < M && nc < N) acc0[t][r] = C[(size_t)m0 * ldc + nc];
                    if (m1 < M && nc < N) acc1[t][r] = C[(size_t)m1 * ldc + nc];
                }
            }
        }
    }

    for (int k0 = 0; k0 < Kp; k0 += 32) {
        __builtin_prefetch(arow0 + k0 + 32, 0, 1);
        const v16bf af0 = cat16(*(const v8bf*)(arow0 + k0), *(const v8bf*)(arow0 + k0 + 16));
        const v16bf af1 = cat16(*(const v8bf*)(arow1 + k0), *(const v8bf*)(arow1 + k0 + 16));

        const v16bf bf0 = cat16(*(const v8bf*)(brow0 + k0), *(const v8bf*)(brow0 + k0 + 16));
        const v16bf bf1 = cat16(*(const v8bf*)(brow1 + k0), *(const v8bf*)(brow1 + k0 + 16));
        const v16bf bf2 = cat16(*(const v8bf*)(brow2 + k0), *(const v8bf*)(brow2 + k0 + 16));
        const v16bf bf3 = cat16(*(const v8bf*)(brow3 + k0), *(const v8bf*)(brow3 + k0 + 16));

        acc0[0] = __builtin_amdgcn_wmma_f32_16x16x32_bf16(false, af0, false, bf0, (short)0, acc0[0], false, false);
        acc1[0] = __builtin_amdgcn_wmma_f32_16x16x32_bf16(false, af1, false, bf0, (short)0, acc1[0], false, false);
        acc0[1] = __builtin_amdgcn_wmma_f32_16x16x32_bf16(false, af0, false, bf1, (short)0, acc0[1], false, false);
        acc1[1] = __builtin_amdgcn_wmma_f32_16x16x32_bf16(false, af1, false, bf1, (short)0, acc1[1], false, false);
        acc0[2] = __builtin_amdgcn_wmma_f32_16x16x32_bf16(false, af0, false, bf2, (short)0, acc0[2], false, false);
        acc1[2] = __builtin_amdgcn_wmma_f32_16x16x32_bf16(false, af1, false, bf2, (short)0, acc1[2], false, false);
        acc0[3] = __builtin_amdgcn_wmma_f32_16x16x32_bf16(false, af0, false, bf3, (short)0, acc0[3], false, false);
        acc1[3] = __builtin_amdgcn_wmma_f32_16x16x32_bf16(false, af1, false, bf3, (short)0, acc1[3], false, false);
    }

    // Epilogue. D layout: m = tile_m + {0,16} + r + 8*hsel, n = tn0 + 16t + lr.
    if (full) {
        #pragma unroll
        for (int t = 0; t < 4; ++t) {
            const int nc = tn0 + 16 * t + lr;
            const float bv = bias ? bias[nc] : 0.0f;
            #pragma unroll
            for (int r = 0; r < 8; ++r) {
                float v0 = acc0[t][r] + bv;
                float v1 = acc1[t][r] + bv;
                if (act == 1) {
                    v0 = (v0 > 20.0f) ? v0 : log1pf(__expf(v0));
                    v1 = (v1 > 20.0f) ? v1 : log1pf(__expf(v1));
                }
                C[(size_t)(tile_m + r + 8 * hsel) * ldc + nc]      = v0;
                C[(size_t)(tile_m + 16 + r + 8 * hsel) * ldc + nc] = v1;
            }
        }
    } else {
        #pragma unroll
        for (int t = 0; t < 4; ++t) {
            const int nc = tn0 + 16 * t + lr;
            const float bv = (bias && nc < N) ? bias[nc] : 0.0f;
            #pragma unroll
            for (int r = 0; r < 8; ++r) {
                const int m0 = tile_m + r + 8 * hsel;
                const int m1 = m0 + 16;
                if (m0 < M && nc < N) {
                    float v = acc0[t][r] + bv;
                    if (act == 1) v = (v > 20.0f) ? v : log1pf(__expf(v));
                    C[(size_t)m0 * ldc + nc] = v;
                }
                if (m1 < M && nc < N) {
                    float v = acc1[t][r] + bv;
                    if (act == 1) v = (v > 20.0f) ? v : log1pf(__expf(v));
                    C[(size_t)m1 * ldc + nc] = v;
                }
            }
        }
    }
}

// ---------------------------------------------------------------------------
// Weight prepass: W (fp32, K x N, ld=ldw) -> Wt (bf16, Npad x Kp, N-major),
// zero-padded in both dims.
// ---------------------------------------------------------------------------
__global__ void wt_bf16_kernel(const float* __restrict__ W, int ldw,
                               __bf16* __restrict__ Wt,
                               int K, int N, int Npad, int Kp)
{
    const int i = blockIdx.x * blockDim.x + threadIdx.x;
    const int total = Npad * Kp;
    if (i >= total) return;
    const int k = i % Kp;
    const int n = i / Kp;
    const float v = (n < N && k < K) ? W[(size_t)k * ldw + n] : 0.0f;
    Wt[i] = (__bf16)v;
}

// ---------------------------------------------------------------------------
// Generic fp32 (M x lda, first K cols) -> bf16 (M x Kp) zero-padded.
// ---------------------------------------------------------------------------
__global__ void abf_kernel(const float* __restrict__ A, int lda,
                           __bf16* __restrict__ Abf, int M, int K, int Kp)
{
    const int i = blockIdx.x * blockDim.x + threadIdx.x;
    if (i >= M * Kp) return;
    const int k = i % Kp;
    const int m = i / Kp;
    Abf[i] = (__bf16)((k < K) ? A[(size_t)m * lda + k] : 0.0f);
}

// ---------------------------------------------------------------------------
// LayerNorm (population variance), one wave per token, bf16 output padded
// to Kp columns (GEMM A operand).
// ---------------------------------------------------------------------------
__global__ void layernorm_bf16_kernel(const float* __restrict__ x,
                                      const float* __restrict__ w,
                                      const float* __restrict__ b,
                                      __bf16* __restrict__ o,
                                      int M, int d, int Kp)
{
    const int tok  = (blockIdx.x * blockDim.x + threadIdx.x) >> 5;
    const int lane = threadIdx.x & 31;
    if (tok >= M) return;
    const float* row = x + (size_t)tok * d;

    float s = 0.0f;
    for (int j = lane; j < d; j += 32) s += row[j];
    #pragma unroll
    for (int off = 16; off > 0; off >>= 1) s += __shfl_xor(s, off, 32);
    const float mu = s / (float)d;

    float v = 0.0f;
    for (int j = lane; j < d; j += 32) { const float u = row[j] - mu; v += u * u; }
    #pragma unroll
    for (int off = 16; off > 0; off >>= 1) v += __shfl_xor(v, off, 32);
    const float rs = rsqrtf(v / (float)d + 1e-5f);

    __bf16* orow = o + (size_t)tok * Kp;
    for (int j = lane; j < Kp; j += 32) {
        float y = 0.0f;
        if (j < d) y = (row[j] - mu) * rs * w[j] + b[j];
        orow[j] = (__bf16)y;
    }
}

// ---------------------------------------------------------------------------
// Depthwise causal conv1d (taps=4) + SiLU.  xi = first di columns of XZ
// (row width 2*di).  Writes fp32 XC (for the scan) and bf16 XCbf (GEMM A).
// ---------------------------------------------------------------------------
__global__ void dwconv_silu_kernel(const float* __restrict__ xz,
                                   const float* __restrict__ w,
                                   const float* __restrict__ b,
                                   float* __restrict__ xc,
                                   __bf16* __restrict__ xcbf,
                                   int Bsz, int L, int di)
{
    const int i = blockIdx.x * blockDim.x + threadIdx.x;
    const int total = Bsz * L * di;
    if (i >= total) return;
    const int ch = i % di;
    const int l  = (i / di) % L;
    const int n  = i / (di * L);
    const int ldx = 2 * di;

    float acc = b[ch];
    #pragma unroll
    for (int k = 0; k < 4; ++k) {
        const int ll = l + k - 3;
        if (ll >= 0)
            acc += w[ch * 4 + k] * xz[((size_t)n * L + ll) * ldx + ch];
    }
    const float y = silu_f(acc);
    const size_t o = ((size_t)n * L + l) * di + ch;
    xc[o]   = y;
    xcbf[o] = (__bf16)y;
}

// ---------------------------------------------------------------------------
// Selective scan: one thread per (batch, channel), 16-wide register state.
// Writes U = (ys + xc*D) * silu(z) directly as bf16 (out_proj GEMM A operand).
// ---------------------------------------------------------------------------
__global__ void scan_kernel(const float* __restrict__ dt,    // (M, di)
                            const float* __restrict__ xc,    // (M, di)
                            const float* __restrict__ dbl,   // (M, dblw); B at +dtr, C at +dtr+16
                            int dtr, int dblw,
                            const float* __restrict__ xz,    // (M, 2di); z at +di
                            const float* __restrict__ A_log, // (di, 16)
                            const float* __restrict__ Dp,    // (di)
                            __bf16* __restrict__ U,          // (M, di) bf16
                            int Bsz, int L, int di)
{
    const int t = blockIdx.x * blockDim.x + threadIdx.x;
    if (t >= Bsz * di) return;
    const int b  = t / di;
    const int ch = t % di;

    float Arow[16];
    #pragma unroll
    for (int s = 0; s < 16; ++s) Arow[s] = -__expf(A_log[ch * 16 + s]);
    float h[16];
    #pragma unroll
    for (int s = 0; s < 16; ++s) h[s] = 0.0f;
    const float Dch = Dp[ch];

    for (int l = 0; l < L; ++l) {
        const size_t tok = (size_t)b * L + l;
        const float dtv = dt[tok * di + ch];
        const float xv  = xc[tok * di + ch];
        const float dx  = dtv * xv;
        const float* Bm = dbl + tok * dblw + dtr;
        const float* Cm = Bm + 16;
        float y = 0.0f;
        #pragma unroll
        for (int s = 0; s < 16; ++s) {
            h[s] = __expf(dtv * Arow[s]) * h[s] + dx * Bm[s];
            y += h[s] * Cm[s];
        }
        const float zv = xz[tok * (2 * di) + di + ch];
        U[tok * di + ch] = (__bf16)((y + xv * Dch) * silu_f(zv));
    }
}

// ---------------------------------------------------------------------------
// GLU gate: x12 (M, 4d) -> bf16 u (M, 2d): u[:,j] = silu(x12[:,j]) * x12[:,2d+j]
// ---------------------------------------------------------------------------
__global__ void glu_kernel(const float* __restrict__ x12,
                           __bf16* __restrict__ u, int M, int d2)
{
    const int i = blockIdx.x * blockDim.x + threadIdx.x;
    if (i >= M * d2) return;
    const int j = i % d2;
    const int t = i / d2;
    const float x1 = x12[(size_t)t * (2 * d2) + j];
    const float x2 = x12[(size_t)t * (2 * d2) + d2 + j];
    u[i] = (__bf16)(silu_f(x1) * x2);
}

// ---------------------------------------------------------------------------
// 3x3 SAME conv (stride 1). in: NCHW (in_blc==0) or (N, H*W, C) token-major
// (in_blc==1). out: NCHW. Optional leaky-relu(0.2).
// ---------------------------------------------------------------------------
__global__ void conv3x3_kernel(const float* __restrict__ in,
                               const float* __restrict__ w,
                               const float* __restrict__ bias,
                               float* __restrict__ out,
                               int N, int IC, int OC, int H, int Wd,
                               int lrelu, int in_blc)
{
    const int i = blockIdx.x * blockDim.x + threadIdx.x;
    const int total = N * OC * H * Wd;
    if (i >= total) return;
    const int x  = i % Wd;
    const int y  = (i / Wd) % H;
    const int oc = (i / (Wd * H)) % OC;
    const int n  = i / (Wd * H * OC);

    float acc = bias[oc];
    for (int ic = 0; ic < IC; ++ic) {
        #pragma unroll
        for (int ky = 0; ky < 3; ++ky) {
            const int iy = y + ky - 1;
            if (iy < 0 || iy >= H) continue;
            #pragma unroll
            for (int kx = 0; kx < 3; ++kx) {
                const int ix = x + kx - 1;
                if (ix < 0 || ix >= Wd) continue;
                const float v = in_blc
                    ? in[((size_t)(n * H + iy) * Wd + ix) * IC + ic]
                    : in[(((size_t)n * IC + ic) * H + iy) * Wd + ix];
                acc += w[((oc * IC + ic) * 3 + ky) * 3 + kx] * v;
            }
        }
    }
    out[i] = lrelu ? (acc >= 0.0f ? acc : 0.2f * acc) : acc;
}

// ---------------------------------------------------------------------------
// Haar DWT of (f1+f2): f1,f2 NCHW (N,C,2H,2W) -> Hi (N,H*W,3C), Lo (N,H*W,C)
// token-major (matches _to3).
// ---------------------------------------------------------------------------
__global__ void dwt_kernel(const float* __restrict__ f1,
                           const float* __restrict__ f2,
                           float* __restrict__ Hi, float* __restrict__ Lo,
                           int N, int C, int H, int Wd)
{
    const int i = blockIdx.x * blockDim.x + threadIdx.x;
    const int total = N * C * H * Wd;
    if (i >= total) return;
    const int x = i % Wd;
    const int y = (i / Wd) % H;
    const int c = (i / (Wd * H)) % C;
    const int n = i / (Wd * H * C);

    const int H2 = 2 * H, W2 = 2 * Wd;
    const size_t base = ((size_t)n * C + c) * H2 * W2;
    const size_t i00 = base + (size_t)(2 * y) * W2 + 2 * x;
    const size_t i10 = i00 + W2;

    const float a  = f1[i00]     + f2[i00];
    const float b  = f1[i00 + 1] + f2[i00 + 1];
    const float cc = f1[i10]     + f2[i10];
    const float d  = f1[i10 + 1] + f2[i10 + 1];

    const size_t tok = (size_t)n * H * Wd + (size_t)y * Wd + x;
    Hi[tok * (3 * C) + c]         = (-a - b + cc + d) * 0.5f;
    Hi[tok * (3 * C) + C + c]     = (-a + b - cc + d) * 0.5f;
    Hi[tok * (3 * C) + 2 * C + c] = ( a - b - cc + d) * 0.5f;
    Lo[tok * C + c]               = ( a + b + cc + d) * 0.5f;
}

// ---------------------------------------------------------------------------
// Inverse DWT: Hi (N,H*W,3C), Lo (N,H*W,C) -> out (N,(2H)*(2W),C) token-major.
// ---------------------------------------------------------------------------
__global__ void idwt_kernel(const float* __restrict__ Hi,
                            const float* __restrict__ Lo,
                            float* __restrict__ out,
                            int N, int C, int H, int Wd)
{
    const int i = blockIdx.x * blockDim.x + threadIdx.x;
    const int total = N * C * H * Wd;
    if (i >= total) return;
    const int x = i % Wd;
    const int y = (i / Wd) % H;
    const int c = (i / (Wd * H)) % C;
    const int n = i / (Wd * H * C);

    const size_t tok = (size_t)n * H * Wd + (size_t)y * Wd + x;
    const float lo = Lo[tok * C + c];
    const float lh = Hi[tok * (3 * C) + c];
    const float hl = Hi[tok * (3 * C) + C + c];
    const float hh = Hi[tok * (3 * C) + 2 * C + c];

    const int H2 = 2 * H, W2 = 2 * Wd;
    const size_t obase = (size_t)n * H2 * W2;
    out[(obase + (size_t)(2 * y)     * W2 + 2 * x)     * C + c] = (lo - lh - hl + hh) * 0.5f;
    out[(obase + (size_t)(2 * y)     * W2 + 2 * x + 1) * C + c] = (lo - lh + hl - hh) * 0.5f;
    out[(obase + (size_t)(2 * y + 1) * W2 + 2 * x)     * C + c] = (lo + lh - hl - hh) * 0.5f;
    out[(obase + (size_t)(2 * y + 1) * W2 + 2 * x + 1) * C + c] = (lo + lh + hl + hh) * 0.5f;
}

// ===========================================================================
// Host side
// ===========================================================================
struct BlockP {
    const float *ln1_w, *ln1_b;
    const float *in_proj, *conv_w, *conv_b, *x_proj, *dt_w, *dt_b, *A_log, *D, *out_proj;
    const float *ln2_w, *ln2_b, *fc1_w, *fc1_b, *fc2_w, *fc2_b;
};

static void read_block(void* const* d_in, int& i, BlockP& p)
{
    p.ln1_w   = (const float*)d_in[i++];
    p.ln1_b   = (const float*)d_in[i++];
    p.in_proj = (const float*)d_in[i++];
    p.conv_w  = (const float*)d_in[i++];
    p.conv_b  = (const float*)d_in[i++];
    p.x_proj  = (const float*)d_in[i++];
    p.dt_w    = (const float*)d_in[i++];
    p.dt_b    = (const float*)d_in[i++];
    p.A_log   = (const float*)d_in[i++];
    p.D       = (const float*)d_in[i++];
    p.out_proj= (const float*)d_in[i++];
    p.ln2_w   = (const float*)d_in[i++];
    p.ln2_b   = (const float*)d_in[i++];
    p.fc1_w   = (const float*)d_in[i++];
    p.fc1_b   = (const float*)d_in[i++];
    p.fc2_w   = (const float*)d_in[i++];
    p.fc2_b   = (const float*)d_in[i++];
}

// Prepares Wt (weight transpose->bf16), then runs the WMMA GEMM.
static void gemm(hipStream_t s, const __bf16* Abf, const float* W, int ldw,
                 __bf16* WT, float* C, int ldc, const float* bias,
                 int M, int N, int K, int act, int accum)
{
    const int Kp = rup32(K), Npad = rup64(N);
    const int TB = 256;
    const int tot = Npad * Kp;
    wt_bf16_kernel<<<(tot + TB - 1) / TB, TB, 0, s>>>(W, ldw, WT, K, N, Npad, Kp);
    dim3 grid(Npad / 64, (M + 127) / 128);
    gemm_wmma_bf16<<<grid, 128, 0, s>>>(Abf, WT, C, ldc, bias, M, N, Kp, act, accum);
}

static void run_block(hipStream_t s, const BlockP& p, float* x,
                      int Bsz, int L, int d,
                      float* XZ, float* XC, float* DBL, float* DT,
                      __bf16* ABF, __bf16* WT)
{
    const int di   = 2 * d;
    int dtr        = d / 16; if (dtr < 1) dtr = 1;
    const int dblw = dtr + 32;
    const int M    = Bsz * L;
    const int TB   = 256;

    // ln1 -> bf16 A operand
    layernorm_bf16_kernel<<<(M * 32 + TB - 1) / TB, TB, 0, s>>>(x, p.ln1_w, p.ln1_b, ABF, M, d, rup32(d));
    gemm(s, ABF, p.in_proj, 2 * di, WT, XZ, 2 * di, nullptr, M, 2 * di, d, 0, 0);
    {
        const int total = Bsz * L * di;
        dwconv_silu_kernel<<<(total + TB - 1) / TB, TB, 0, s>>>(XZ, p.conv_w, p.conv_b, XC, ABF, Bsz, L, di);
    }
    gemm(s, ABF, p.x_proj, dblw, WT, DBL, dblw, nullptr, M, dblw, di, 0, 0);
    {
        const int total = M * 32;   // Kp(dtr)=32
        abf_kernel<<<(total + TB - 1) / TB, TB, 0, s>>>(DBL, dblw, ABF, M, dtr, 32);
    }
    gemm(s, ABF, p.dt_w, di, WT, DT, di, p.dt_b, M, di, dtr, /*softplus*/1, 0);
    {
        const int total = Bsz * di;
        scan_kernel<<<(total + TB - 1) / TB, TB, 0, s>>>(DT, XC, DBL, dtr, dblw, XZ, p.A_log, p.D, ABF, Bsz, L, di);
    }
    gemm(s, ABF, p.out_proj, d, WT, x, d, nullptr, M, d, di, 0, /*residual*/1);

    // MLP
    layernorm_bf16_kernel<<<(M * 32 + TB - 1) / TB, TB, 0, s>>>(x, p.ln2_w, p.ln2_b, ABF, M, d, rup32(d));
    gemm(s, ABF, p.fc1_w, 4 * d, WT, XZ /*X12*/, 4 * d, p.fc1_b, M, 4 * d, d, 0, 0);
    {
        const int total = M * 2 * d;
        glu_kernel<<<(total + TB - 1) / TB, TB, 0, s>>>(XZ, ABF, M, 2 * d);
    }
    gemm(s, ABF, p.fc2_w, d, WT, x, d, p.fc2_b, M, d, 2 * d, 0, /*residual*/1);
}

extern "C" void kernel_launch(void* const* d_in, const int* in_sizes, int n_in,
                              void* d_out, int out_size, void* d_ws, size_t ws_size,
                              hipStream_t stream)
{
    (void)in_sizes; (void)n_in; (void)out_size; (void)ws_size;

    int i = 0;
    const float* Ain = (const float*)d_in[i++];   // (2,1,128,128)
    const float* Bin = (const float*)d_in[i++];   // (2,1,128,128)
    const float* stem_w = (const float*)d_in[i++];
    const float* stem_b = (const float*)d_in[i++];
    BlockP hi[2], lo[2], fu[4];
    for (int k = 0; k < 2; ++k) read_block(d_in, i, hi[k]);
    for (int k = 0; k < 2; ++k) read_block(d_in, i, lo[k]);
    for (int k = 0; k < 4; ++k) read_block(d_in, i, fu[k]);
    const float* l1w = (const float*)d_in[i++];
    const float* l1b = (const float*)d_in[i++];
    const float* l2w = (const float*)d_in[i++];
    const float* l2b = (const float*)d_in[i++];
    const float* l3w = (const float*)d_in[i++];
    const float* l3b = (const float*)d_in[i++];

    // ---------------- workspace layout (float units) ----------------
    float* ws = (float*)d_ws;
    const size_t O_F1    = 0;                   // 2*64*128*128 = 2097152
    const size_t O_F2    = O_F1    + 2097152;
    const size_t O_XHI   = O_F2    + 2097152;   // 2*4096*192 = 1572864
    const size_t O_XLO   = O_XHI   + 1572864;   // 2*4096*64  = 524288
    const size_t O_XFUSE = O_XLO   + 524288;    // 2*16384*64 = 2097152
    const size_t O_XZ    = O_XFUSE + 2097152;   // max 8388608 (aliases X12)
    const size_t O_XC    = O_XZ    + 8388608;   // max 4194304
    const size_t O_DBL   = O_XC    + 4194304;   // max 1441792
    const size_t O_DT    = O_DBL   + 1441792;   // max 4194304
    const size_t O_ABF   = O_DT    + 4194304;   // 4194304 bf16 = 2097152 floats
    const size_t O_WT    = O_ABF   + 2097152;   // 196608 bf16 = 98304 floats

    float*  F1    = ws + O_F1;
    float*  F2    = ws + O_F2;
    float*  XHI   = ws + O_XHI;
    float*  XLO   = ws + O_XLO;
    float*  XFUSE = ws + O_XFUSE;
    float*  XZ    = ws + O_XZ;
    float*  XC    = ws + O_XC;
    float*  DBL   = ws + O_DBL;
    float*  DT    = ws + O_DT;
    __bf16* ABF   = (__bf16*)(ws + O_ABF);
    __bf16* WT    = (__bf16*)(ws + O_WT);

    const int TB = 256;

    // ---- stem conv + lrelu on both inputs (1 -> 64 ch, 128x128)
    {
        const int total = 2 * 64 * 128 * 128;
        conv3x3_kernel<<<(total + TB - 1) / TB, TB, 0, stream>>>(
            Ain, stem_w, stem_b, F1, 2, 1, 64, 128, 128, 1, 0);
        conv3x3_kernel<<<(total + TB - 1) / TB, TB, 0, stream>>>(
            Bin, stem_w, stem_b, F2, 2, 1, 64, 128, 128, 1, 0);
    }

    // ---- DWT (fused sum) -> XHI (B,4096,192), XLO (B,4096,64)
    {
        const int total = 2 * 64 * 64 * 64;
        dwt_kernel<<<(total + TB - 1) / TB, TB, 0, stream>>>(F1, F2, XHI, XLO, 2, 64, 64, 64);
    }

    // ---- hi blocks (d=192, L=4096), lo blocks (d=64, L=4096)
    for (int k = 0; k < 2; ++k)
        run_block(stream, hi[k], XHI, 2, 4096, 192, XZ, XC, DBL, DT, ABF, WT);
    for (int k = 0; k < 2; ++k)
        run_block(stream, lo[k], XLO, 2, 4096, 64, XZ, XC, DBL, DT, ABF, WT);

    // ---- IDWT -> XFUSE (B, 16384, 64) token-major
    {
        const int total = 2 * 64 * 64 * 64;
        idwt_kernel<<<(total + TB - 1) / TB, TB, 0, stream>>>(XHI, XLO, XFUSE, 2, 64, 64, 64);
    }

    // ---- fuse blocks (d=64, L=16384)
    for (int k = 0; k < 4; ++k)
        run_block(stream, fu[k], XFUSE, 2, 16384, 64, XZ, XC, DBL, DT, ABF, WT);

    // ---- tail convs: 64->32 (lrelu), 32->16 (lrelu), 16->1
    {
        float* C1 = F1;  // reuse stem buffers
        float* C2 = F2;
        int total = 2 * 32 * 128 * 128;
        conv3x3_kernel<<<(total + TB - 1) / TB, TB, 0, stream>>>(
            XFUSE, l1w, l1b, C1, 2, 64, 32, 128, 128, 1, /*in_blc=*/1);
        total = 2 * 16 * 128 * 128;
        conv3x3_kernel<<<(total + TB - 1) / TB, TB, 0, stream>>>(
            C1, l2w, l2b, C2, 2, 32, 16, 128, 128, 1, 0);
        total = 2 * 1 * 128 * 128;
        conv3x3_kernel<<<(total + TB - 1) / TB, TB, 0, stream>>>(
            C2, l3w, l3b, (float*)d_out, 2, 16, 1, 128, 128, 0, 0);
    }
}
